// MPSCompatibleMultiheadAttention_9010841387693
// MI455X (gfx1250) — matrix operand order repaired and checked by reading, not compiled, not run on hardware
//
#include <hip/hip_runtime.h>
#include <hip/hip_bf16.h>

// ---------------- types ----------------
typedef __bf16 bf16_t;
typedef __attribute__((ext_vector_type(16))) __bf16 v16bf;
typedef __attribute__((ext_vector_type(8)))  __bf16 v8bf;
typedef __attribute__((ext_vector_type(4)))  __bf16 v4bf;
typedef __attribute__((ext_vector_type(8)))  float  v8f;
typedef __attribute__((ext_vector_type(4)))  int    v4i;

#define WMMA_BF16(a, b, c) \
  __builtin_amdgcn_wmma_f32_16x16x32_bf16(false, (a), false, (b), (short)0, (c), false, false)

__device__ __forceinline__ v16bf cat8(v8bf lo, v8bf hi) {
  return __builtin_shufflevector(lo, hi, 0,1,2,3,4,5,6,7,8,9,10,11,12,13,14,15);
}

// problem constants
#define BATCH 4
#define SEQ   2048
#define EMB   1024
#define HEADS 16
#define HDIM  64
#define NTOK  (BATCH * SEQ)   // 8192

// ---------------- async global->LDS path (guarded; sync fallback) ----------------
#if defined(__gfx1250__) && __has_builtin(__builtin_amdgcn_global_load_async_to_lds_b128)
#define HAVE_ASYNC_LDS 1
#else
#define HAVE_ASYNC_LDS 0
#endif

#if HAVE_ASYNC_LDS
#if __has_builtin(__builtin_amdgcn_s_wait_asynccnt)
#define WAIT_ASYNC() __builtin_amdgcn_s_wait_asynccnt(0)
#else
#define WAIT_ASYNC() asm volatile("s_wait_asynccnt 0" ::: "memory")
#endif
typedef __attribute__((address_space(1))) v4i* gv4i_p;  // global int4*
typedef __attribute__((address_space(3))) v4i* lv4i_p;  // LDS int4*
__device__ __forceinline__ void async_b128(const void* g, void* l) {
  __builtin_amdgcn_global_load_async_to_lds_b128(
      (gv4i_p)(uintptr_t)g,
      (lv4i_p)(uint32_t)(uintptr_t)l,
      0, 0);
}
#else
#define WAIT_ASYNC() ((void)0)
#endif

// ---------------- fp32 -> bf16 convert (x4 vectorized) ----------------
__global__ void cvt_f32_bf16(const float* __restrict__ src, bf16_t* __restrict__ dst, int n4) {
  int i = blockIdx.x * 256 + threadIdx.x;
  if (i >= n4) return;
  float4 f = ((const float4*)src)[i];
  v4bf o;
  o[0] = (bf16_t)f.x; o[1] = (bf16_t)f.y; o[2] = (bf16_t)f.z; o[3] = (bf16_t)f.w;
  ((v4bf*)dst)[i] = o;
}

// ---------------- projection GEMM: C[m,n] = sum_k A[m,k] * W[n,k] + bias[n] ----------------
// Double-buffered (K=64 stage) LDS tiles fed by async global->LDS DMA.
// MODE 0: bf16 out, layout [b,h,s,d]      (Q, K)
// MODE 1: bf16 out, layout [b,h,d,s]      (V transposed)
// MODE 2: f32  out, layout [m,n] -> d_out (output projection)
template<int MODE>
__global__ __launch_bounds__(256)
void gemm_wmma(const bf16_t* __restrict__ A, const bf16_t* __restrict__ W,
               const float* __restrict__ bias, void* __restrict__ Cout) {
  __shared__ __align__(16) bf16_t At[2][64][64];    // 8 KB x2
  __shared__ __align__(16) bf16_t Bt[2][128][64];   // 16 KB x2

  const int tid  = threadIdx.x;
  const int lane = tid & 31;
  const int wave = tid >> 5;
  const int lr = lane & 15;     // tile row/col selector
  const int lg = lane >> 4;     // half-wave group
  const int wm = wave & 3;      // 4 waves along M
  const int wn = wave >> 2;     // 2 waves along N
  const int m0blk = blockIdx.y * 64;
  const int n0blk = blockIdx.x * 128;

  // tile-load assignments per K=64 stage: A: 2x b128/thread, B: 4x b128/thread
  const int arow = tid >> 2;            // 0..63
  const int acol = (tid & 3) * 16;      // 0,16,32,48
  const int brow = tid >> 1;            // 0..127
  const int bcol = (tid & 1) * 32;      // 0,32

  const bf16_t* gA = A + (size_t)(m0blk + arow) * EMB + acol;
  const bf16_t* gB = W + (size_t)(n0blk + brow) * EMB + bcol;

  auto load_tiles = [&](int buf, int kk) {
    bf16_t* la = &At[buf][arow][acol];
    bf16_t* lb = &Bt[buf][brow][bcol];
#if HAVE_ASYNC_LDS
    async_b128(gA + kk,      la);
    async_b128(gA + kk + 8,  la + 8);
    async_b128(gB + kk,      lb);
    async_b128(gB + kk + 8,  lb + 8);
    async_b128(gB + kk + 16, lb + 16);
    async_b128(gB + kk + 24, lb + 24);
#else
    *(v8bf*)la        = *(const v8bf*)(gA + kk);
    *(v8bf*)(la + 8)  = *(const v8bf*)(gA + kk + 8);
    *(v8bf*)lb        = *(const v8bf*)(gB + kk);
    *(v8bf*)(lb + 8)  = *(const v8bf*)(gB + kk + 8);
    *(v8bf*)(lb + 16) = *(const v8bf*)(gB + kk + 16);
    *(v8bf*)(lb + 24) = *(const v8bf*)(gB + kk + 24);
#endif
  };

  v8f acc[4] = {};

  load_tiles(0, 0);
  WAIT_ASYNC();
  __syncthreads();

  int buf = 0;
  for (int kk = 0; kk < EMB; kk += 64) {
    if (kk + 64 < EMB) load_tiles(buf ^ 1, kk + 64);

    const bf16_t* ar = &At[buf][wm * 16 + lr][0];
#pragma unroll
    for (int kc = 0; kc < 2; ++kc) {
      const v16bf af = cat8(*(const v8bf*)(ar + kc * 32 + lg * 8),
                            *(const v8bf*)(ar + kc * 32 + 16 + lg * 8));
#pragma unroll
      for (int i = 0; i < 4; ++i) {
        const bf16_t* br = &Bt[buf][wn * 64 + i * 16 + lr][kc * 32 + lg * 8];
        const v16bf bfr = cat8(*(const v8bf*)(br), *(const v8bf*)(br + 16));
        acc[i] = WMMA_BF16(af, bfr, acc[i]);
      }
    }

    WAIT_ASYNC();
    __syncthreads();
    buf ^= 1;
  }

#pragma unroll
  for (int i = 0; i < 4; ++i) {
    const int n = n0blk + wn * 64 + i * 16 + lr;
    const float bv = bias[n];
#pragma unroll
    for (int r = 0; r < 8; ++r) {
      const int m = m0blk + wm * 16 + r + lg * 8;
      const float v = acc[i][r] + bv;
      if (MODE == 2) {
        ((float*)Cout)[(size_t)m * EMB + n] = v;
      } else {
        const int b = m >> 11, s = m & (SEQ - 1);
        const int h = n >> 6,  d = n & (HDIM - 1);
        if (MODE == 0)
          ((bf16_t*)Cout)[(((size_t)b * HEADS + h) * SEQ + s) * HDIM + d] = (bf16_t)v;
        else
          ((bf16_t*)Cout)[(((size_t)b * HEADS + h) * HDIM + d) * SEQ + s] = (bf16_t)v;
      }
    }
  }
}

// ---------------- fused causal attention ----------------
// grid: (SEQ/16, BATCH); block: 256 threads (8 waves).
// Per workgroup: 16 query rows, loops all 16 heads; scores + head-average in LDS.
__global__ __launch_bounds__(256, 1)
void attn_kernel(const bf16_t* __restrict__ Q, const bf16_t* __restrict__ K,
                 const bf16_t* __restrict__ Vt, bf16_t* __restrict__ attn,
                 float* __restrict__ avg_out) {
  __shared__ float sc[16][SEQ + 4];   // per-head score / prob rows
  __shared__ float av[16][SEQ + 4];   // head-averaged probs

  const int b  = blockIdx.y;
  const int q0 = blockIdx.x * 16;
  const int lane = threadIdx.x & 31;
  const int wave = threadIdx.x >> 5;
  const int lr = lane & 15;
  const int lg = lane >> 4;

  for (int idx = threadIdx.x; idx < 16 * SEQ; idx += 256)
    av[idx >> 11][idx & (SEQ - 1)] = 0.0f;
  __syncthreads();

  for (int h = 0; h < HEADS; ++h) {
    const bf16_t* Qh = Q  + (((size_t)b * HEADS + h) * SEQ) * HDIM;
    const bf16_t* Kh = K  + (((size_t)b * HEADS + h) * SEQ) * HDIM;
    const bf16_t* Vh = Vt + (((size_t)b * HEADS + h) * HDIM) * SEQ;

    // ---- phase A: scores = Q K^T / sqrt(d), causal masked ----
    const bf16_t* qrowp = Qh + (size_t)(q0 + lr) * HDIM;
    const v16bf aq0 = cat8(*(const v8bf*)(qrowp + lg * 8),
                           *(const v8bf*)(qrowp + 16 + lg * 8));
    const v16bf aq1 = cat8(*(const v8bf*)(qrowp + 32 + lg * 8),
                           *(const v8bf*)(qrowp + 48 + lg * 8));
    for (int t = 0; t < 16; ++t) {
      const int j0 = (wave + t * 8) * 16;
      const bf16_t* krow = Kh + (size_t)(j0 + lr) * HDIM;
      if (t + 1 < 16)   // WGP-scope prefetch of next K tile for this wave
        __builtin_prefetch(Kh + (size_t)(j0 + 128 + lr) * HDIM, 0, 3);
      const v16bf bk0 = cat8(*(const v8bf*)(krow + lg * 8),
                             *(const v8bf*)(krow + 16 + lg * 8));
      const v16bf bk1 = cat8(*(const v8bf*)(krow + 32 + lg * 8),
                             *(const v8bf*)(krow + 48 + lg * 8));
      v8f s = {};
      s = WMMA_BF16(aq0, bk0, s);
      s = WMMA_BF16(aq1, bk1, s);
#pragma unroll
      for (int r = 0; r < 8; ++r) {
        const int row = r + lg * 8;
        const int col = j0 + lr;
        float v = s[r] * 0.125f;                 // 1/sqrt(64)
        if (col > q0 + row) v = -1e30f;          // causal mask
        sc[row][col] = v;
      }
    }
    __syncthreads();

    // ---- phase B: row softmax (wave per 2 rows), accumulate head average ----
#pragma unroll
    for (int rr = 0; rr < 2; ++rr) {
      const int r = wave * 2 + rr;
      float m = -1e30f;
      for (int i = lane; i < SEQ; i += 32) m = fmaxf(m, sc[r][i]);
#pragma unroll
      for (int o = 16; o; o >>= 1) m = fmaxf(m, __shfl_xor(m, o, 32));
      float ssum = 0.0f;
      for (int i = lane; i < SEQ; i += 32) ssum += __expf(sc[r][i] - m);
#pragma unroll
      for (int o = 16; o; o >>= 1) ssum += __shfl_xor(ssum, o, 32);
      const float inv = 1.0f / ssum;
      for (int i = lane; i < SEQ; i += 32) {
        const float p = __expf(sc[r][i] - m) * inv;
        sc[r][i] = p;
        av[r][i] += p * (1.0f / HEADS);
      }
    }
    __syncthreads();

    // ---- phase C: O = P V  (waves 0-3: first half of keys, 4-7: second half) ----
    const int n0   = (wave & 3) * 16;
    const int half = wave >> 2;
    v8f o = {};
    for (int cj = half * 32; cj < half * 32 + 32; ++cj) {
      const int jb = cj * 32;
      v16bf ap;
      const float* srow = &sc[lr][0];
#pragma unroll
      for (int e = 0; e < 8; ++e) ap[e]     = (bf16_t)srow[jb + lg * 8 + e];
#pragma unroll
      for (int e = 0; e < 8; ++e) ap[8 + e] = (bf16_t)srow[jb + 16 + lg * 8 + e];
      const bf16_t* vrow = Vh + (size_t)(n0 + lr) * SEQ + jb + lg * 8;
      const v16bf bv = cat8(*(const v8bf*)(vrow), *(const v8bf*)(vrow + 16));
      o = WMMA_BF16(ap, bv, o);
    }
    __syncthreads();
    float* red = &sc[0][0];                     // reuse score LDS for cross-wave reduce
    if (half == 1) {
#pragma unroll
      for (int r = 0; r < 8; ++r)
        red[(wave & 3) * 256 + (r + lg * 8) * 16 + lr] = o[r];
    }
    __syncthreads();
    if (half == 0) {
#pragma unroll
      for (int r = 0; r < 8; ++r) {
        const float v = o[r] + red[(wave & 3) * 256 + (r + lg * 8) * 16 + lr];
        const int q = q0 + r + lg * 8;
        attn[((size_t)b * SEQ + q) * EMB + h * HDIM + n0 + lr] = (bf16_t)v;
      }
    }
    __syncthreads();
  }

  // ---- write head-averaged attention weights ----
  for (int idx = threadIdx.x; idx < 16 * SEQ; idx += 256) {
    const int r = idx >> 11, c = idx & (SEQ - 1);
    avg_out[((size_t)b * SEQ + (q0 + r)) * SEQ + c] = av[r][c];
  }
}

// ---------------- host launcher ----------------
extern "C" void kernel_launch(void* const* d_in, const int* in_sizes, int n_in,
                              void* d_out, int out_size, void* d_ws, size_t ws_size,
                              hipStream_t stream) {
  (void)in_sizes; (void)n_in; (void)out_size; (void)ws_size;
  const float* query = (const float*)d_in[0];
  const float* key   = (const float*)d_in[1];
  const float* value = (const float*)d_in[2];
  // d_in[3] = attn_mask (causal, implicit)
  const float* wq = (const float*)d_in[4];  const float* bq = (const float*)d_in[5];
  const float* wk = (const float*)d_in[6];  const float* bk = (const float*)d_in[7];
  const float* wv = (const float*)d_in[8];  const float* bv = (const float*)d_in[9];
  const float* wo = (const float*)d_in[10]; const float* bo = (const float*)d_in[11];

  const size_t NE = (size_t)NTOK * EMB;     // 8,388,608
  const size_t WE = (size_t)EMB * EMB;      // 1,048,576
  bf16_t* w = (bf16_t*)d_ws;
  bf16_t* Xq = w;            bf16_t* Xk = w + NE;       bf16_t* Xv = w + 2 * NE;
  bf16_t* Wq = w + 3 * NE;   bf16_t* Wk = Wq + WE;      bf16_t* Wv = Wk + WE;  bf16_t* Wo = Wv + WE;
  bf16_t* Qb = Wo + WE;      bf16_t* Kb = Qb + NE;      bf16_t* Vtb = Kb + NE;
  bf16_t* attn = Xq;                         // Xq dead after Q projection -> reuse

  float* out_attn = (float*)d_out;
  float* out_avg  = out_attn + NE;

  // 1) fp32 -> bf16
  const int gA = (int)(NE / 4 / 256), gW = (int)(WE / 4 / 256);
  cvt_f32_bf16<<<gA, 256, 0, stream>>>(query, Xq, (int)(NE / 4));
  cvt_f32_bf16<<<gA, 256, 0, stream>>>(key,   Xk, (int)(NE / 4));
  cvt_f32_bf16<<<gA, 256, 0, stream>>>(value, Xv, (int)(NE / 4));
  cvt_f32_bf16<<<gW, 256, 0, stream>>>(wq, Wq, (int)(WE / 4));
  cvt_f32_bf16<<<gW, 256, 0, stream>>>(wk, Wk, (int)(WE / 4));
  cvt_f32_bf16<<<gW, 256, 0, stream>>>(wv, Wv, (int)(WE / 4));
  cvt_f32_bf16<<<gW, 256, 0, stream>>>(wo, Wo, (int)(WE / 4));

  // 2) projections
  dim3 ggrid(EMB / 128, NTOK / 64);
  gemm_wmma<0><<<ggrid, 256, 0, stream>>>(Xq, Wq, bq, Qb);
  gemm_wmma<0><<<ggrid, 256, 0, stream>>>(Xk, Wk, bk, Kb);
  gemm_wmma<1><<<ggrid, 256, 0, stream>>>(Xv, Wv, bv, Vtb);

  // 3) fused causal attention + head-averaged weights
  attn_kernel<<<dim3(SEQ / 16, BATCH), 256, 0, stream>>>(Qb, Kb, Vtb, attn, out_avg);

  // 4) output projection -> fp32 d_out
  gemm_wmma<2><<<ggrid, 256, 0, stream>>>(attn, Wo, bo, out_attn);
}
